// FlashAttention_75943611728136
// MI455X (gfx1250) — matrix operand order repaired and checked by reading, not compiled, not run on hardware
//
#include <hip/hip_runtime.h>

typedef __attribute__((ext_vector_type(16))) __bf16 v16bf;
typedef __attribute__((ext_vector_type(8)))  __bf16 v8bf;
typedef __attribute__((ext_vector_type(8)))  float  v8f;

#define B_  2
#define T_  2048
#define D_  2048
#define H_  16
#define HD_ 128
#define SCALE_ 0.022097086912079608f   // 2048^-0.5 (reference scales by full embed_dim)

__device__ __forceinline__ __bf16 f2bf(float f) {
  unsigned u = __builtin_bit_cast(unsigned, f);
  unsigned r = u + 0x7FFFu + ((u >> 16) & 1u);      // round-to-nearest-even
  unsigned short h = (unsigned short)(r >> 16);
  return __builtin_bit_cast(__bf16, h);
}

__device__ __forceinline__ v16bf cat16(v8bf lo, v8bf hi) {
  v16bf r;
#pragma unroll
  for (int i = 0; i < 8; ++i) { r[i] = lo[i]; r[i + 8] = hi[i]; }
  return r;
}

// ---- CDNA5 async global->LDS DMA (ASYNCcnt-tracked), GVS addressing:
// mem = SGPR_base + VGPR_voff ; LDS[lds_voff] <- 16B per lane per op.
__device__ __forceinline__ void async_copy64B(unsigned lds, const char* sbase,
                                              unsigned voff) {
#pragma unroll
  for (int j = 0; j < 4; ++j)
    asm volatile("global_load_async_to_lds_b128 %0, %1, %2 offset:0"
                 :: "v"(lds + 16u * j), "v"(voff + 16u * j), "s"(sbase)
                 : "memory");
}

__device__ __forceinline__ void async_copy128B(unsigned lds, const char* sbase,
                                               unsigned voff) {
#pragma unroll
  for (int j = 0; j < 8; ++j)
    asm volatile("global_load_async_to_lds_b128 %0, %1, %2 offset:0"
                 :: "v"(lds + 16u * j), "v"(voff + 16u * j), "s"(sbase)
                 : "memory");
}

__device__ __forceinline__ void wait_async0() {
  asm volatile("s_wait_asynccnt 0x0" ::: "memory");
}

__device__ __forceinline__ unsigned lds_off(const void* p) {
  return (unsigned)(size_t)p;   // low 32 bits of generic ptr = LDS offset
}

// ---------------------------------------------------------------- casts
__global__ __launch_bounds__(256) void cast_bf16_kernel(
    const float* __restrict__ in, __bf16* __restrict__ out, int n) {
  int i = (blockIdx.x * blockDim.x + threadIdx.x) * 4;
  if (i + 3 < n) {
    float4 v = *(const float4*)(in + i);
    out[i + 0] = f2bf(v.x);
    out[i + 1] = f2bf(v.y);
    out[i + 2] = f2bf(v.z);
    out[i + 3] = f2bf(v.w);
  }
}

// W[K][N] (fp32) -> Wt[N][K] (bf16), 32x32 LDS-tiled transpose
__global__ __launch_bounds__(256) void transpose_cast_kernel(
    const float* __restrict__ W, __bf16* __restrict__ Wt) {
  __shared__ float tile[32][33];
  int n0 = blockIdx.x * 32, k0 = blockIdx.y * 32;
  int tx = threadIdx.x & 31, ty = threadIdx.x >> 5;   // ty: 0..7
#pragma unroll
  for (int i = ty; i < 32; i += 8)
    tile[i][tx] = W[(size_t)(k0 + i) * D_ + (n0 + tx)];
  __syncthreads();
#pragma unroll
  for (int i = ty; i < 32; i += 8)
    Wt[(size_t)(n0 + i) * D_ + (k0 + tx)] = f2bf(tile[tx][i]);
}

// ---------------------------------------------------------------- GEMM
// C = A[M x K] * Wt^T  (Wt stored [N][K] row-major), bf16 in, f32 accum.
// Double-buffered LDS fed by async global->LDS DMA.
// MODE 0: out bf16 at [b][h][t][hd]     (Q / K layout)
// MODE 1: out bf16 at [b][h][hd][t]     (V transposed layout)
// MODE 2: out fp32 row-major [m][n] + bias (final projection)
#define GT_M 128
#define GT_N 128
#define GT_K 64
#define GST  80    // 64 + 16 pad; 160B row stride, multiple of 16B

template <int MODE>
__global__ __launch_bounds__(256) void gemm_bf16_kernel(
    const __bf16* __restrict__ A, const __bf16* __restrict__ Bt,
    const float* __restrict__ bias, void* __restrict__ out,
    int M, int N, int K, float oscale) {
  __shared__ __bf16 As[2][GT_M * GST];
  __shared__ __bf16 Bs[2][GT_N * GST];
  const int tid  = threadIdx.x;
  const int lane = tid & 31;
  const int w    = tid >> 5;
  const int half = lane >> 4, nl = lane & 15;
  const int m0 = blockIdx.x * GT_M, n0 = blockIdx.y * GT_N;
  const int wm = (w & 1) * 64, wn = (w >> 1) * 32;
  v8f acc[4][2] = {};
  const int lr = tid >> 1, lc = (tid & 1) * 32;

  const char* baseA = (const char*)(A  + (size_t)m0 * K);
  const char* baseB = (const char*)(Bt + (size_t)n0 * K);
  const unsigned rowoff = (unsigned)((lr * K + lc) * 2);
  unsigned ldsA[2], ldsB[2];
#pragma unroll
  for (int i = 0; i < 2; ++i) {
    ldsA[i] = lds_off(&As[i][lr * GST + lc]);
    ldsB[i] = lds_off(&Bs[i][lr * GST + lc]);
  }

  // prologue: DMA first K-tile into buffer 0
  async_copy64B(ldsA[0], baseA, rowoff);
  async_copy64B(ldsB[0], baseB, rowoff);
  wait_async0();
  __syncthreads();

  const int nk = K / GT_K;
  for (int it = 0; it < nk; ++it) {
    const int cur = it & 1;
    if (it + 1 < nk) {   // DMA next tile into other buffer, overlapped w/ WMMA
      const unsigned koff = rowoff + (unsigned)((it + 1) * GT_K * 2);
      async_copy64B(ldsA[cur ^ 1], baseA, koff);
      async_copy64B(ldsB[cur ^ 1], baseB, koff);
    }
#pragma unroll
    for (int kc = 0; kc < GT_K; kc += 32) {
      v16bf af[4], bf2[2];
#pragma unroll
      for (int mi = 0; mi < 4; ++mi) {
        const __bf16* p = &As[cur][(wm + mi * 16 + nl) * GST + kc + half * 8];
        af[mi] = cat16(*(const v8bf*)p, *(const v8bf*)(p + 16));
      }
#pragma unroll
      for (int ni = 0; ni < 2; ++ni) {
        const __bf16* p = &Bs[cur][(wn + ni * 16 + nl) * GST + kc + half * 8];
        bf2[ni] = cat16(*(const v8bf*)p, *(const v8bf*)(p + 16));
      }
#pragma unroll
      for (int mi = 0; mi < 4; ++mi)
#pragma unroll
        for (int ni = 0; ni < 2; ++ni)
          acc[mi][ni] = __builtin_amdgcn_wmma_f32_16x16x32_bf16(
              false, af[mi], false, bf2[ni], (short)0, acc[mi][ni], false, false);
    }
    wait_async0();
    __syncthreads();
  }

  // epilogue: C layout -> VGPR r holds row M = r + 8*half, col N = lane%16
#pragma unroll
  for (int mi = 0; mi < 4; ++mi)
#pragma unroll
    for (int ni = 0; ni < 2; ++ni)
#pragma unroll
      for (int r = 0; r < 8; ++r) {
        int m = m0 + wm + mi * 16 + r + 8 * half;
        int n = n0 + wn + ni * 16 + nl;
        float v = (acc[mi][ni][r] + bias[n]) * oscale;
        if (MODE == 2) {
          ((float*)out)[(size_t)m * N + n] = v;
        } else {
          int b = m >> 11, t = m & (T_ - 1);
          int h = n >> 7,  hd = n & (HD_ - 1);
          size_t idx = (MODE == 0)
              ? (((size_t)b * H_ + h) * T_ + t) * HD_ + hd
              : (((size_t)b * H_ + h) * HD_ + hd) * T_ + t;
          ((__bf16*)out)[idx] = f2bf(v);
        }
      }
}

// ---------------------------------------------------------------- attention
// One block per (b,h, 128-query tile); wave w owns q-rows [16w, 16w+16).
// K/V^T tiles double-buffered via async DMA; Q pre-scaled by SCALE_.
#define AST 136   // 128 + 8 pad; 272B row stride, multiple of 16B

__global__ __launch_bounds__(256) void attn_kernel(
    const __bf16* __restrict__ Q, const __bf16* __restrict__ Kd,
    const __bf16* __restrict__ Vt, __bf16* __restrict__ O) {
  __shared__ __bf16 Qs[128 * AST];
  __shared__ __bf16 Ks[2][128 * AST];
  __shared__ __bf16 Vs[2][128 * AST];     // V^T tile: [dim][key]
  __shared__ __bf16 Ps[8 * 16 * AST];     // per-wave P strip [16 rows][128 keys]
  const int tid = threadIdx.x, lane = tid & 31, w = tid >> 5;
  const int half = lane >> 4, nl = lane & 15;
  const int q0 = blockIdx.x * 128;
  const int bh = blockIdx.y;
  const size_t hoff = (size_t)bh * T_ * HD_;

  const int cr = tid >> 1, cc = (tid & 1) * 64;   // cooperative copy coords
  const unsigned ldsQ = lds_off(&Qs[cr * AST + cc]);
  unsigned ldsK[2], ldsV[2];
#pragma unroll
  for (int i = 0; i < 2; ++i) {
    ldsK[i] = lds_off(&Ks[i][cr * AST + cc]);
    ldsV[i] = lds_off(&Vs[i][cr * AST + cc]);
  }
  const char* qbase = (const char*)(Q  + hoff + (size_t)q0 * HD_);
  const char* kbase = (const char*)(Kd + hoff);
  const char* vbase = (const char*)(Vt + hoff);

  // prologue DMA: Q tile + first K/V^T tiles into buffer 0
  async_copy128B(ldsQ,    qbase, (unsigned)((cr * HD_ + cc) * 2));
  async_copy128B(ldsK[0], kbase, (unsigned)((cr * HD_ + cc) * 2));
  async_copy128B(ldsV[0], vbase, (unsigned)((cr * T_  + cc) * 2));
  wait_async0();
  __syncthreads();

  v8f o[8] = {};
  float mr[8], ls[8];
#pragma unroll
  for (int r = 0; r < 8; ++r) { mr[r] = -1e30f; ls[r] = 0.f; }
  __bf16* pw = &Ps[w * 16 * AST];

  const int nkt = T_ / 128;
  for (int it = 0; it < nkt; ++it) {
    const int cur = it & 1;
    if (it + 1 < nkt) {   // DMA next K/V tile, overlapped with compute
      const int kt = (it + 1) * 128;
      async_copy128B(ldsK[cur ^ 1], kbase, (unsigned)(((kt + cr) * HD_ + cc) * 2));
      async_copy128B(ldsV[cur ^ 1], vbase, (unsigned)((cr * T_ + kt + cc) * 2));
    }

    // S = Q K^T  (contract over head dim, 4 chunks of 32)
    v8f s[8] = {};
#pragma unroll
    for (int kc = 0; kc < 128; kc += 32) {
      v16bf aq;
      { const __bf16* p = &Qs[(w * 16 + nl) * AST + kc + half * 8];
        aq = cat16(*(const v8bf*)p, *(const v8bf*)(p + 16)); }
      v16bf bk[8];
#pragma unroll
      for (int nt = 0; nt < 8; ++nt) {
        const __bf16* p = &Ks[cur][(nt * 16 + nl) * AST + kc + half * 8];
        bk[nt] = cat16(*(const v8bf*)p, *(const v8bf*)(p + 16));
      }
#pragma unroll
      for (int nt = 0; nt < 8; ++nt)
        s[nt] = __builtin_amdgcn_wmma_f32_16x16x32_bf16(
            false, aq, false, bk[nt], (short)0, s[nt], false, false);
    }

    // online softmax (row = r + 8*half; 16 cols of a tile live across 16 lanes)
    float mloc[8];
#pragma unroll
    for (int r = 0; r < 8; ++r) {
      float mx = -1e30f;
#pragma unroll
      for (int nt = 0; nt < 8; ++nt) mx = fmaxf(mx, s[nt][r]);
#pragma unroll
      for (int d = 1; d < 16; d <<= 1)
        mx = fmaxf(mx, __shfl_xor(mx, d, 32));
      mloc[r] = mx;
    }
    float al[8], mn[8], sl[8];
#pragma unroll
    for (int r = 0; r < 8; ++r) {
      mn[r] = fmaxf(mr[r], mloc[r]);
      al[r] = __expf(mr[r] - mn[r]);
      mr[r] = mn[r];
      sl[r] = 0.f;
    }
#pragma unroll
    for (int nt = 0; nt < 8; ++nt)
#pragma unroll
      for (int r = 0; r < 8; ++r) {
        o[nt][r] *= al[r];
        float p = __expf(s[nt][r] - mn[r]);
        s[nt][r] = p;
        sl[r] += p;
      }
#pragma unroll
    for (int r = 0; r < 8; ++r) {
      float t = sl[r];
#pragma unroll
      for (int d = 1; d < 16; d <<= 1)
        t += __shfl_xor(t, d, 32);
      ls[r] = ls[r] * al[r] + t;
    }

    // P (C-layout) -> per-wave LDS strip as A-layout source
#pragma unroll
    for (int nt = 0; nt < 8; ++nt)
#pragma unroll
      for (int r = 0; r < 8; ++r)
        pw[(r + 8 * half) * AST + nt * 16 + nl] = f2bf(s[nt][r]);

    // O += P @ V (contract over 128 keys)
#pragma unroll
    for (int kc = 0; kc < 128; kc += 32) {
      v16bf ap;
      { const __bf16* p = &pw[nl * AST + kc + half * 8];
        ap = cat16(*(const v8bf*)p, *(const v8bf*)(p + 16)); }
      v16bf bv[8];
#pragma unroll
      for (int nt = 0; nt < 8; ++nt) {
        const __bf16* p = &Vs[cur][(nt * 16 + nl) * AST + kc + half * 8];
        bv[nt] = cat16(*(const v8bf*)p, *(const v8bf*)(p + 16));
      }
#pragma unroll
      for (int nt = 0; nt < 8; ++nt)
        o[nt] = __builtin_amdgcn_wmma_f32_16x16x32_bf16(
            false, ap, false, bv[nt], (short)0, o[nt], false, false);
    }

    wait_async0();        // next tile's DMA landed
    __syncthreads();      // and everyone is done reading buffer `cur`
  }

  // normalize + write attn output in [b][t][d] layout (bf16)
  const int b = bh >> 4, h = bh & (H_ - 1);
#pragma unroll
  for (int r = 0; r < 8; ++r) {
    float inv = 1.0f / ls[r];
    int t = q0 + w * 16 + r + 8 * half;
#pragma unroll
    for (int nt = 0; nt < 8; ++nt) {
      int d = h * HD_ + nt * 16 + nl;
      O[((size_t)b * T_ + t) * D_ + d] = f2bf(o[nt][r] * inv);
    }
  }
}

// ---------------------------------------------------------------- launch
extern "C" void kernel_launch(void* const* d_in, const int* in_sizes, int n_in,
                              void* d_out, int out_size, void* d_ws, size_t ws_size,
                              hipStream_t stream) {
  const float* x  = (const float*)d_in[0];
  const float* wq = (const float*)d_in[1];
  const float* bq = (const float*)d_in[2];
  const float* wk = (const float*)d_in[3];
  const float* bk = (const float*)d_in[4];
  const float* wv = (const float*)d_in[5];
  const float* bv = (const float*)d_in[6];
  const float* wo = (const float*)d_in[7];
  const float* bo = (const float*)d_in[8];
  float* out = (float*)d_out;

  char* ws = (char*)d_ws;
  const size_t XB = (size_t)B_ * T_ * D_ * 2;   // 16 MiB (bf16 activations)
  const size_t WT = (size_t)D_ * D_ * 2;        //  8 MiB (bf16 weight)
  __bf16* xb  = (__bf16*)(ws);
  __bf16* wqt = (__bf16*)(ws + XB);
  __bf16* wkt = (__bf16*)(ws + XB + 1 * WT);
  __bf16* wvt = (__bf16*)(ws + XB + 2 * WT);
  __bf16* wot = (__bf16*)(ws + XB + 3 * WT);
  __bf16* Qb  = (__bf16*)(ws + 1 * XB + 4 * WT);
  __bf16* Kb  = (__bf16*)(ws + 2 * XB + 4 * WT);
  __bf16* Vtb = (__bf16*)(ws + 3 * XB + 4 * WT);
  __bf16* Ab  = (__bf16*)(ws + 4 * XB + 4 * WT);

  int nx = B_ * T_ * D_;
  cast_bf16_kernel<<<nx / 1024, 256, 0, stream>>>(x, xb, nx);
  dim3 tg(D_ / 32, D_ / 32);
  transpose_cast_kernel<<<tg, 256, 0, stream>>>(wq, wqt);
  transpose_cast_kernel<<<tg, 256, 0, stream>>>(wk, wkt);
  transpose_cast_kernel<<<tg, 256, 0, stream>>>(wv, wvt);
  transpose_cast_kernel<<<tg, 256, 0, stream>>>(wo, wot);

  dim3 gg((B_ * T_) / GT_M, D_ / GT_N);
  gemm_bf16_kernel<0><<<gg, 256, 0, stream>>>(xb, wqt, bq, Qb,  B_ * T_, D_, D_, SCALE_);
  gemm_bf16_kernel<0><<<gg, 256, 0, stream>>>(xb, wkt, bk, Kb,  B_ * T_, D_, D_, 1.0f);
  gemm_bf16_kernel<1><<<gg, 256, 0, stream>>>(xb, wvt, bv, Vtb, B_ * T_, D_, D_, 1.0f);

  dim3 ag(T_ / 128, B_ * H_);
  attn_kernel<<<ag, 256, 0, stream>>>(Qb, Kb, Vtb, Ab);

  gemm_bf16_kernel<2><<<gg, 256, 0, stream>>>(Ab, wot, bo, out, B_ * T_, D_, D_, 1.0f);

  (void)in_sizes; (void)n_in; (void)out_size; (void)ws_size;
}
